// TaskAlignedAssigner_23124103922076
// MI455X (gfx1250) — compile-verified, matrix operand order checked by reading, not run on hardware
//
#include <hip/hip_runtime.h>
#include <hip/hip_bf16.h>
#include <stdint.h>

// Problem constants (match reference)
#define Bn 64
#define An 8400
#define Cn 80
#define Mn 32
#define KTOP 13
#define NSLOT (Mn * KTOP)   // 416 candidate slots per batch

// ---------------------------------------------------------------------------
// CIoU between gt (g*) and pred (p*), xyxy, eps = 1e-7 (matches reference _ciou)
// ---------------------------------------------------------------------------
__device__ __forceinline__ float ciou_f(float gx1, float gy1, float gx2, float gy2,
                                        float px1, float py1, float px2, float py2)
{
    const float eps = 1e-7f;
    float w1 = gx2 - gx1, h1 = gy2 - gy1;
    float w2 = px2 - px1, h2 = py2 - py1;
    float iw = fmaxf(fminf(gx2, px2) - fmaxf(gx1, px1), 0.0f);
    float ih = fmaxf(fminf(gy2, py2) - fmaxf(gy1, py1), 0.0f);
    float inter = iw * ih;
    float uni   = w1 * h1 + w2 * h2 - inter + eps;
    float iou   = inter / uni;
    float cw = fmaxf(gx2, px2) - fminf(gx1, px1);
    float ch = fmaxf(gy2, py2) - fminf(gy1, py1);
    float c2 = cw * cw + ch * ch + eps;
    float dx = px1 + px2 - gx1 - gx2;
    float dy = py1 + py2 - gy1 - gy2;
    float rho2 = (dx * dx + dy * dy) * 0.25f;
    float at = atanf(w2 / (h2 + eps)) - atanf(w1 / (h1 + eps));
    float v  = 0.4052847345693511f * at * at;   // 4/pi^2
    float alpha = v / (v - iou + (1.0f + eps));
    return iou - (rho2 / c2 + v * alpha);
}

// anchors_mask: anchor point strictly inside gt box (min delta > 1e-9)
__device__ __forceinline__ bool in_box(float ax, float ay,
                                       float gx1, float gy1, float gx2, float gy2)
{
    float d0 = ax - gx1, d1 = ay - gy1, d2 = gx2 - ax, d3 = gy2 - ay;
    float dmin = fminf(fminf(d0, d1), fminf(d2, d3));
    return dmin > 1e-9f;
}

// ---------------------------------------------------------------------------
// Kernel 1: one block per (b,m). Compute align[a] = score * iou^6 (masked) into
// LDS, then 13 rounds of stable argmax (lowest index wins ties, matching
// jax.lax.top_k ordering). Writes 13 indices per (b,m) to workspace;
// -1 sentinel row for gt_mask==0 (reference forces those idx->0 which makes
// counts==13 at anchor 0 -> topk_mask all false, i.e. contributes nothing).
// ---------------------------------------------------------------------------
__global__ __launch_bounds__(256) void k1_topk(
    const float* __restrict__ pd_scores, const float* __restrict__ pd_boxes,
    const float* __restrict__ anchors,   const int* __restrict__ gt_classes,
    const float* __restrict__ gt_boxes,  const unsigned char* __restrict__ gt_mask,
    int* __restrict__ topk_idx)
{
    __shared__ float sAl[An];        // 33.6 KB: align for all anchors
    __shared__ float sRv[256];
    __shared__ int   sRi[256];
    __shared__ int   sWin[KTOP];

    const int bm  = blockIdx.x;      // bm = b*32 + m
    const int b   = bm >> 5;
    const int tid = threadIdx.x;

    if (gt_mask[bm] == 0) {          // masked gt: inert sentinel row
        if (tid < KTOP) topk_idx[bm * KTOP + tid] = -1;
        return;
    }

    const float gx1 = gt_boxes[bm * 4 + 0], gy1 = gt_boxes[bm * 4 + 1];
    const float gx2 = gt_boxes[bm * 4 + 2], gy2 = gt_boxes[bm * 4 + 3];
    const int   cls = gt_classes[bm];

    const float4* pb4 = (const float4*)pd_boxes + (size_t)b * An;
    const float*  sc  = pd_scores + (size_t)b * An * Cn + cls;

    for (int a = tid; a < An; a += 256) {
        // CDNA5 prefetch path (global_prefetch_b8); speculative, OOB-safe
        __builtin_prefetch(pb4 + a + 256, 0, 1);
        float4 pb = pb4[a];
        float ax = anchors[2 * a + 0], ay = anchors[2 * a + 1];
        float al = 0.0f;
        if (in_box(ax, ay, gx1, gy1, gx2, gy2)) {
            float score = sc[(size_t)a * Cn];                      // gather class column
            float iou = fmaxf(ciou_f(gx1, gy1, gx2, gy2, pb.x, pb.y, pb.z, pb.w), 0.0f);
            float i2 = iou * iou;
            al = score * (i2 * i2 * i2);                           // ALPHA=1, BETA=6
        }
        sAl[a] = al;                                               // align >= 0
    }
    __syncthreads();

    for (int r = 0; r < KTOP; ++r) {
        float bv = -1.0f; int bi = An;
        for (int a = tid; a < An; a += 256) {                      // ascending: strict >
            float v = sAl[a];                                      // keeps lowest index
            if (v > bv) { bv = v; bi = a; }
        }
        sRv[tid] = bv; sRi[tid] = bi;
        __syncthreads();
        for (int s = 128; s > 0; s >>= 1) {
            if (tid < s) {
                float v2 = sRv[tid + s]; int j2 = sRi[tid + s];
                if (v2 > sRv[tid] || (v2 == sRv[tid] && j2 < sRi[tid])) {
                    sRv[tid] = v2; sRi[tid] = j2;
                }
            }
            __syncthreads();
        }
        if (tid == 0) { int w = sRi[0]; sWin[r] = w; sAl[w] = -1.0f; }
        __syncthreads();
    }
    if (tid < KTOP) topk_idx[bm * KTOP + tid] = sWin[tid];
}

// ---------------------------------------------------------------------------
// Kernel 3: default outputs. target_scores=0, fg=0, target_boxes=gt_boxes[b,0]
// (reference: argmax over all-zero positive gives index 0). B128 stores.
// d_out layout: [B*A*C scores][B*A*4 boxes][B*A fg], all float32.
// ---------------------------------------------------------------------------
__global__ __launch_bounds__(256) void k3_defaults(
    float* __restrict__ out, const float* __restrict__ gt_boxes)
{
    const size_t N1 = (size_t)Bn * An * Cn;     // 43,008,000
    const size_t NB = (size_t)Bn * An;          //    537,600
    const size_t stride = (size_t)gridDim.x * blockDim.x;
    const size_t i0 = (size_t)blockIdx.x * blockDim.x + threadIdx.x;
    const float4 z = make_float4(0.f, 0.f, 0.f, 0.f);

    float4* o4 = (float4*)out;
    for (size_t i = i0; i < N1 / 4; i += stride) o4[i] = z;        // scores = 0

    float4* ob = (float4*)(out + N1);
    const float4* gb4 = (const float4*)gt_boxes;
    for (size_t i = i0; i < NB; i += stride) {
        int b = (int)(i / An);
        ob[i] = gb4[(size_t)b * Mn];                               // gt_boxes[b,0,:]
    }

    float4* of = (float4*)(out + N1 + NB * 4);
    for (size_t i = i0; i < NB / 4; i += stride) of[i] = z;        // fg = 0
}

// ---------------------------------------------------------------------------
// Kernel 2: one block per batch. Duplicate resolution + norm + scatter.
//  * Tensor Data Mover (tensor_load_to_lds + s_wait_tensorcnt) stages
//    gt_boxes[b] into LDS (plain-load overwrite keeps it correct regardless
//    of unverified TDM descriptor semantics).
//  * V_WMMA_F32_16X16X32_F16 computes the duplicate counts:
//      count[s] = sum_t pp[t] * (anchor[t] == anchor[s]),  t over 416 slots,
//    as D = A x ones + C with a 0/1 f16 A-matrix (exact: sums <= 32).
//    A is built branchlessly from sPPA[t] = pp[t] ? anchor[t] : -2
//    (sentinel -2 never equals any myAnc, including -1 of empty slots).
// ---------------------------------------------------------------------------
typedef unsigned int u32x4 __attribute__((ext_vector_type(4)));
typedef int          i32x4 __attribute__((ext_vector_type(4)));
typedef int          i32x8 __attribute__((ext_vector_type(8)));
typedef __attribute__((ext_vector_type(16))) _Float16 v16h;
typedef __attribute__((ext_vector_type(8)))  float    v8f;

__global__ __launch_bounds__(256) void k2_resolve(
    const float* __restrict__ pd_scores, const float* __restrict__ pd_boxes,
    const float* __restrict__ anchors,   const int* __restrict__ gt_classes,
    const float* __restrict__ gt_boxes,  const int* __restrict__ topk_idx,
    float* __restrict__ out)
{
    __shared__ float    sGtb[Mn * 4];   // gt boxes (TDM target)
    __shared__ int      sGtc[Mn];
    __shared__ int      sAnc[NSLOT];    // candidate anchor per slot (-1 = none)
    __shared__ int      sPP[NSLOT];     // pre-positive (mask & topk & gt_mask)
    __shared__ int      sPPA[NSLOT];    // pp ? anchor : -2  (branchless WMMA feed)
    __shared__ int      sPm[NSLOT];     // resolved m* per owned positive anchor
    __shared__ float    sPa[NSLOT];     // align at (m*, a)
    __shared__ int      sCnt[NSLOT];    // WMMA duplicate counts per slot
    __shared__ int      sClaim[An];     // owner slot per candidate anchor (33.6 KB)
    __shared__ unsigned sMaxA[Mn];      // per-m max align (uint bits, vals >= 0)
    __shared__ unsigned sMaxI[Mn];      // per-m max iou

    const int b   = blockIdx.x;
    const int tid = threadIdx.x;

    // --- CDNA5 TDM async tensor load: gt_boxes[b] -> LDS (128 f32 = 512 B) ---
    if (tid < 32) {  // issue from wave 0 only (TENSOR ops ignore EXEC)
        unsigned lds_off = (unsigned)(size_t)(&sGtb[0]);           // LDS aperture low bits
        unsigned long long ga =
            (unsigned long long)(size_t)(gt_boxes + (size_t)b * Mn * 4);
        u32x4 g0 = {0u, 0u, 0u, 0u};
        i32x8 g1 = {0, 0, 0, 0, 0, 0, 0, 0};
        i32x4 g2 = {0, 0, 0, 0};
        i32x4 g3 = {0, 0, 0, 0};
        // D# group 0: count=1, lds_addr, global_addr[56:0], type=2 ("image")
        g0[0] = 1u;
        g0[1] = lds_off;
        g0[2] = (unsigned)(ga & 0xFFFFFFFFull);
        g0[3] = (unsigned)((ga >> 32) & 0x1FFFFFFull) | (2u << 30);
        // D# group 1: data_size=2 (4B), tensor_dim0=128, tensor_dim1=1,
        //             tile_dim0=128, tile_dim1=1, tile_dim2=0, dim0_stride=128
        g1[0] = (2 << 16);
        g1[1] = (int)(128u << 16);      // tensor_dim0 low16 at bits 63:48
        g1[2] = (int)(1u << 16);        // tensor_dim0 hi=0 | tensor_dim1 lo=1
        g1[3] = (int)(128u << 16);      // tensor_dim1 hi=0 | tile_dim0=128
        g1[4] = 1;                      // tile_dim1=1 | tile_dim2=0
        g1[5] = 128;                    // tensor_dim0_stride low32
        g1[6] = 0;
        g1[7] = 0;
#if defined(__clang_major__) && (__clang_major__ >= 23)
        i32x8 g4 = {0, 0, 0, 0, 0, 0, 0, 0};
        __builtin_amdgcn_tensor_load_to_lds(g0, g1, g2, g3, g4, 0);
#else
        __builtin_amdgcn_tensor_load_to_lds(g0, g1, g2, g3, 0);
#endif
        __builtin_amdgcn_s_wait_tensorcnt(0);
    }
    if (tid < Mn) { sGtc[tid] = gt_classes[b * Mn + tid]; sMaxA[tid] = 0u; sMaxI[tid] = 0u; }
    __syncthreads();
    // Overwrite with plain loads: result correct regardless of TDM semantics.
    if (tid < Mn * 4) sGtb[tid] = gt_boxes[(size_t)b * Mn * 4 + tid];
    __syncthreads();

    // Build candidate slots; pre-positive = (valid gt) & (in topk) & anchors_mask.
    for (int s = tid; s < NSLOT; s += 256) {
        int m = s / KTOP, k = s - m * KTOP;
        int a = topk_idx[(b * Mn + m) * KTOP + k];
        sAnc[s] = a;
        int pp = 0;
        if (a >= 0) {
            float ax = anchors[2 * a], ay = anchors[2 * a + 1];
            pp = in_box(ax, ay, sGtb[m*4+0], sGtb[m*4+1], sGtb[m*4+2], sGtb[m*4+3]) ? 1 : 0;
        }
        sPP[s]  = pp;
        sPPA[s] = pp ? a : -2;
        sPm[s]  = -1;
    }
    __syncthreads();

    // Init claim entries for candidate anchors only.
    for (int s = tid; s < NSLOT; s += 256) {
        int a = sAnc[s];
        if (a >= 0) sClaim[a] = 0x7FFFFFFF;
    }
    __syncthreads();

    // Ownership (lowest pre-positive slot per anchor) via LDS atomicMin.
    for (int s = tid; s < NSLOT; s += 256)
        if (sPP[s]) atomicMin(&sClaim[sAnc[s]], s);

    // Duplicate counts via WMMA (exact; every pre-positive (m,a) pair IS a
    // slot since topk membership is required, so slot-sum == reference counts).
    {
        const int lane  = tid & 31;
        const int wave  = tid >> 5;                 // 8 waves per block
        const int row   = lane & 15;                // A-matrix row M = lane&15
        const int kbase = (lane >= 16) ? 8 : 0;     // 16-bit A 16x32 VGPR layout
        v16h ones;
        for (int i = 0; i < 16; ++i) ones[i] = (_Float16)1.0f;
        for (int chunk = wave; chunk < NSLOT / 16; chunk += 8) {   // 26 row-chunks
            const int myAnc = sAnc[chunk * 16 + row];
            v8f acc = {0.f, 0.f, 0.f, 0.f, 0.f, 0.f, 0.f, 0.f};
            for (int kc = 0; kc < NSLOT / 32; ++kc) {              // 13 K-chunks of 32
                const int t0 = kc * 32;
                v16h a;
                #pragma unroll
                for (int i = 0; i < 16; ++i) {
                    int t = t0 + kbase + ((i < 8) ? i : (i + 8));  // K index per layout
                    // branchless: single compare + select (sentinel -2 never matches)
                    a[i] = (sPPA[t] == myAnc) ? (_Float16)1.0f : (_Float16)0.0f;
                }
                acc = __builtin_amdgcn_wmma_f32_16x16x32_f16(
                          false, a, false, ones, (short)0, acc, false, false);
            }
            if ((lane & 15) == 0) {      // N==0 column: lane0 -> M=0..7, lane16 -> M=8..15
                int mrow = (lane >= 16) ? 8 : 0;
                #pragma unroll
                for (int r = 0; r < 8; ++r)
                    sCnt[chunk * 16 + mrow + r] = (int)(acc[r] + 0.5f);
            }
        }
    }
    __syncthreads();

    // Dup anchors resolved by first-max argmax over ALL 32 gts of masked iou
    // (masked gts participate, per reference semantics).
    for (int s = tid; s < NSLOT; s += 256) {
        if (!sPP[s]) continue;
        const int a = sAnc[s];
        if (sClaim[a] != s) continue;               // owner slot only
        const int m   = s / KTOP;
        const int cnt = sCnt[s];

        float4 pb = ((const float4*)pd_boxes)[(size_t)b * An + a];
        float ax = anchors[2 * a], ay = anchors[2 * a + 1];
        int mS = m;
        if (cnt > 1) {
            float best = -1.0f; int bi = 0;
            for (int mm = 0; mm < Mn; ++mm) {
                float x1 = sGtb[mm*4+0], y1 = sGtb[mm*4+1];
                float x2 = sGtb[mm*4+2], y2 = sGtb[mm*4+3];
                float iou = 0.0f;
                if (in_box(ax, ay, x1, y1, x2, y2))
                    iou = fmaxf(ciou_f(x1, y1, x2, y2, pb.x, pb.y, pb.z, pb.w), 0.0f);
                if (iou > best) { best = iou; bi = mm; }           // first max (jnp.argmax)
            }
            mS = bi;
        }
        // align/iou at (m*, a) (m* may differ from the slot's own m after resolution)
        float x1 = sGtb[mS*4+0], y1 = sGtb[mS*4+1], x2 = sGtb[mS*4+2], y2 = sGtb[mS*4+3];
        float iou = 0.0f, score = 0.0f;
        if (in_box(ax, ay, x1, y1, x2, y2)) {
            iou   = fmaxf(ciou_f(x1, y1, x2, y2, pb.x, pb.y, pb.z, pb.w), 0.0f);
            score = pd_scores[((size_t)b * An + a) * Cn + sGtc[mS]];
        }
        float i2 = iou * iou;
        float al = score * (i2 * i2 * i2);
        sPm[s] = mS;
        sPa[s] = al;
        // nonnegative floats: uint bit-compare is order-preserving
        atomicMax(&sMaxA[mS], __float_as_uint(al));
        atomicMax(&sMaxI[mS], __float_as_uint(iou));
    }
    __syncthreads();

    // Scatter: norm = align * max_iou[m*] / (max_align[m*] + EPS); one-hot class;
    // boxes from gt_boxes[b, m*]; fg = 1. Each anchor has at most one positive m.
    const size_t N1 = (size_t)Bn * An * Cn;
    const size_t NB = (size_t)Bn * An;
    for (int s = tid; s < NSLOT; s += 256) {
        int mS = sPm[s];
        if (mS < 0) continue;
        int a = sAnc[s];
        float maxA = __uint_as_float(sMaxA[mS]);
        float maxI = __uint_as_float(sMaxI[mS]);
        float norm = sPa[s] * maxI / (maxA + 1e-9f);
        int tcls = sGtc[mS]; if (tcls < 0) tcls = 0;
        out[((size_t)b * An + a) * Cn + tcls] = norm;
        ((float4*)(out + N1))[(size_t)b * An + a] =
            make_float4(sGtb[mS*4+0], sGtb[mS*4+1], sGtb[mS*4+2], sGtb[mS*4+3]);
        out[N1 + NB * 4 + (size_t)b * An + a] = 1.0f;
    }
}

// ---------------------------------------------------------------------------
// Host launcher. Inputs (setup_inputs order): pd_scores f32, pd_boxes f32,
// anchor_tensor f32, gt_classes i32, gt_boxes f32, gt_mask bool (1 byte/elem).
// Workspace: only B*M*13 ints (~106 KB) for topk indices.
// ---------------------------------------------------------------------------
extern "C" void kernel_launch(void* const* d_in, const int* in_sizes, int n_in,
                              void* d_out, int out_size, void* d_ws, size_t ws_size,
                              hipStream_t stream)
{
    (void)in_sizes; (void)n_in; (void)out_size; (void)ws_size;
    const float*         pd_scores  = (const float*)d_in[0];
    const float*         pd_boxes   = (const float*)d_in[1];
    const float*         anchors    = (const float*)d_in[2];
    const int*           gt_classes = (const int*)d_in[3];
    const float*         gt_boxes   = (const float*)d_in[4];
    const unsigned char* gt_mask    = (const unsigned char*)d_in[5];
    float* out  = (float*)d_out;
    int*   topk = (int*)d_ws;

    k1_topk<<<Bn * Mn, 256, 0, stream>>>(pd_scores, pd_boxes, anchors,
                                         gt_classes, gt_boxes, gt_mask, topk);
    k3_defaults<<<4096, 256, 0, stream>>>(out, gt_boxes);
    k2_resolve<<<Bn, 256, 0, stream>>>(pd_scores, pd_boxes, anchors,
                                       gt_classes, gt_boxes, topk, out);
}
